// MultiHeadAttention_58755152609358
// MI455X (gfx1250) — compile-verified, matrix operand order checked by reading, not run on hardware
//
#include <hip/hip_runtime.h>
#include <hip/hip_bf16.h>
#include <math.h>

// ---------------------------------------------------------------------------
// MHA block for MI455X (gfx1250, wave32, WMMA f32_16x16x32_f16)
// dims: B=4, S=2048, D=1024, H=16, d=64
// ---------------------------------------------------------------------------

typedef __attribute__((ext_vector_type(16))) _Float16 v16h;
typedef __attribute__((ext_vector_type(8)))  _Float16 v8h;
typedef __attribute__((ext_vector_type(8)))  float    v8f;
typedef __attribute__((ext_vector_type(4)))  int      v4i;

#define B_   4
#define S_   2048
#define D_   1024
#define H_   16
#define DK_  64

#define WMMA_F16(A, Bf, C) \
  __builtin_amdgcn_wmma_f32_16x16x32_f16(false, (A), false, (Bf), (short)0, (C), false, false)

// gfx1250 async global->LDS path (ASYNCcnt), guarded so either branch compiles
#if defined(__has_builtin)
#if __has_builtin(__builtin_amdgcn_global_load_async_to_lds_b128) && \
    __has_builtin(__builtin_amdgcn_s_wait_asynccnt)
#define USE_ASYNC_LDS 1
#endif
#endif
#ifndef USE_ASYNC_LDS
#define USE_ASYNC_LDS 0
#endif

#if USE_ASYNC_LDS
// exact parameter types per hipcc diagnostic: v4i in AS1 (global) / AS3 (LDS)
using gv4i_p = __attribute__((address_space(1))) v4i*;
using lv4i_p = __attribute__((address_space(3))) v4i*;
#endif

// ---- fragment loaders ------------------------------------------------------
// A-frag (and B-frag from a transposed matrix): lane l holds row (l&15) of the
// tile, 16 consecutive K-elements starting at k0 + (l>>4)*16.  (this code's k order, shared by A and B)
__device__ __forceinline__ v16h frag_f16(const _Float16* base, size_t ld,
                                         int row0, int k0, int lane) {
  const _Float16* p = base + (size_t)(row0 + (lane & 15)) * ld
                           + (size_t)(k0 + ((lane >> 4) << 4));
  return *(const v16h*)p;
}

__device__ __forceinline__ v16h frag_from_f32(const float* base, size_t ld,
                                              int row0, int k0, int lane) {
  const float* p = base + (size_t)(row0 + (lane & 15)) * ld
                        + (size_t)(k0 + ((lane >> 4) << 4));
  v16h a;
#pragma unroll
  for (int i = 0; i < 16; ++i) a[i] = (_Float16)p[i];
  return a;
}

// ---------------------------------------------------------------------------
// Kernel W: convert + transpose a 1024x1024 fp32 weight to f16 [N][K]
// ---------------------------------------------------------------------------
__global__ void prep_w_kernel(const float* __restrict__ w, _Float16* __restrict__ wT) {
  int idx = blockIdx.x * 256 + threadIdx.x;          // 1M elements
  int n = idx >> 10;
  int k = idx & 1023;
  wT[((size_t)n << 10) + k] = (_Float16)w[((size_t)k << 10) + n];
}

// ---------------------------------------------------------------------------
// Kernel A: head projection  out = act(f32)[8192x1024] @ W(f16,[N][K]) + bias
// Weight slab (64 rows x 128 K-halves = 16 KB) staged in LDS, double buffered,
// via async global->LDS when available; shared by all 4 waves of the block.
// Output scattered into per-head layout:
//   q/k: [b][h][s][dk]   (stride_s=64, stride_f=1)
//   v  : [b][h][dk][s]   (stride_s=1,  stride_f=S)  -> B-frag ready for attn@V
// block = 128 threads (4 waves); wave = 16 rows x 64 cols; grid=(128,16)
// ---------------------------------------------------------------------------
__global__ void gemm_qkv_kernel(const float* __restrict__ act,
                                const _Float16* __restrict__ wT,
                                const float* __restrict__ bias,
                                _Float16* __restrict__ outp,
                                int stride_s, int stride_f) {
  __shared__ _Float16 bslab[2][64 * 128];            // 2 x 16 KB
  const int t    = threadIdx.x;
  const int w    = t >> 5;
  const int lane = t & 31;
  const int m0 = (blockIdx.x * 4 + w) * 16;          // row tile (b,s flattened)
  const int n0 = blockIdx.y * 64;                    // one head worth of cols

  // stage weight slab rows [n0..n0+63], K [k0..k0+127]; 128B per thread
  auto stage = [&](int buf, int k0) {
    const int row = t >> 1;
    const int off = (t & 1) * 64;                    // halves within row
    const _Float16* g = wT + (size_t)(n0 + row) * D_ + (k0 + off);
    _Float16* l = &bslab[buf][row * 128 + off];
#if USE_ASYNC_LDS
#pragma unroll
    for (int i = 0; i < 8; ++i)
      __builtin_amdgcn_global_load_async_to_lds_b128(
          (gv4i_p)(g + i * 8), (lv4i_p)(l + i * 8), 0, 0);
#else
#pragma unroll
    for (int i = 0; i < 8; ++i)
      *(v8h*)(l + i * 8) = *(const v8h*)(g + i * 8);
#endif
  };

  v8f acc[4] = {};
  stage(0, 0);
  v16h a_next = frag_from_f32(act, D_, m0, 0, lane);

  for (int ko = 0; ko < 8; ++ko) {                   // 8 slabs of 128 K
#if USE_ASYNC_LDS
    __builtin_amdgcn_s_wait_asynccnt(0);
#endif
    __syncthreads();                                 // slab[ko&1] visible
    if (ko + 1 < 8) stage((ko + 1) & 1, (ko + 1) * 128);
    const _Float16* bs = bslab[ko & 1];
#pragma unroll
    for (int ki = 0; ki < 4; ++ki) {
      const v16h a = a_next;
      const int knext = ko * 128 + ki * 32 + 32;
      if (knext < D_) a_next = frag_from_f32(act, D_, m0, knext, lane);
#pragma unroll
      for (int j = 0; j < 4; ++j) {
        v16h bf = frag_f16(bs, 128, j * 16, ki * 32, lane);
        acc[j] = WMMA_F16(a, bf, acc[j]);
      }
    }
    __syncthreads();                                 // done reading slab[ko&1]
  }

  const int half  = lane >> 4;
  const int b     = m0 >> 11;          // tile lives in one batch (16 | 2048)
  const int sbase = m0 & (S_ - 1);
#pragma unroll
  for (int j = 0; j < 4; ++j) {
    const int n  = n0 + j * 16 + (lane & 15);
    const int h  = n >> 6;
    const int dk = n & 63;
    const float bv = bias[n];
    _Float16* hb = outp + (size_t)(b * H_ + h) * S_ * DK_;
#pragma unroll
    for (int r = 0; r < 8; ++r) {
      const int s = sbase + r + 8 * half;
      hb[(size_t)s * stride_s + (size_t)dk * stride_f] = (_Float16)(acc[j][r] + bv);
    }
  }
}

// ---------------------------------------------------------------------------
// Kernel B: attention for one (b,h) and a 16-query tile.
// block = 128 threads (4 waves).  grid = (S/16, B*H).
// LDS: probs32[16][2052] f32 | probs16[16][2064] f16 | reductions  (~198 KB)
// ---------------------------------------------------------------------------
#define PS32 2052
#define PS16 2064
#define SM_P32 (16 * PS32 * 4)
#define SM_P16 (16 * PS16 * 2)
#define SMEM_B (SM_P32 + SM_P16 + (64 + 128 + 16 + 16) * 4)

__global__ void attn_kernel(const _Float16* __restrict__ qh,
                            const _Float16* __restrict__ kh,
                            const _Float16* __restrict__ vt,
                            const int* __restrict__ mask,
                            float* __restrict__ attn_out,
                            _Float16* __restrict__ ctx) {
  extern __shared__ char smem[];
  float*    probs32 = (float*)smem;                          // [16][PS32]
  _Float16* probs16 = (_Float16*)(smem + SM_P32);            // [16][PS16]
  float*    redMax  = (float*)(smem + SM_P32 + SM_P16);      // [16][4]
  float*    redSum  = redMax + 64;                           // [16][8]
  float*    rowMax  = redSum + 128;                          // [16]
  float*    rowInv  = rowMax + 16;                           // [16]

  const int bh = blockIdx.y;
  const int b  = bh >> 4;
  const int s0 = blockIdx.x * 16;
  const int w    = threadIdx.x >> 5;
  const int lane = threadIdx.x & 31;
  const int half = lane >> 4;

  const _Float16* qbase = qh + ((size_t)bh * S_ + s0) * DK_;
  const _Float16* kbase = kh + (size_t)bh * S_ * DK_;
  const _Float16* vbase = vt + (size_t)bh * DK_ * S_;

  // Q tile fragments (reused across all key tiles)
  const v16h aq0 = frag_f16(qbase, DK_, 0, 0, lane);
  const v16h aq1 = frag_f16(qbase, DK_, 0, 32, lane);

  float runMax[8];
#pragma unroll
  for (int r = 0; r < 8; ++r) runMax[r] = -1e30f;

  const float scale = 0.125f;                       // 1/sqrt(64)

  // ---- scores = Q K^T / sqrt(d), masked, into LDS (pipelined K frags) ----
  v16h bk0 = frag_f16(kbase, DK_, w * 512, 0, lane);
  v16h bk1 = frag_f16(kbase, DK_, w * 512, 32, lane);
  for (int kt = 0; kt < 32; ++kt) {
    const int key0 = w * 512 + kt * 16;
    v16h nb0 = bk0, nb1 = bk1;
    if (kt + 1 < 32) {
      nb0 = frag_f16(kbase, DK_, key0 + 16, 0, lane);
      nb1 = frag_f16(kbase, DK_, key0 + 16, 32, lane);
    }
    v8f c = {};
    c = WMMA_F16(aq0, bk0, c);
    c = WMMA_F16(aq1, bk1, c);
    const int col = key0 + (lane & 15);
#pragma unroll
    for (int r = 0; r < 8; ++r) {
      const int m = r + 8 * half;
      float sc = c[r] * scale;
      if (mask[((size_t)b * S_ + (s0 + m)) * S_ + col] == 0) sc = -1e9f;
      probs32[m * PS32 + col] = sc;
      runMax[r] = fmaxf(runMax[r], sc);
    }
    bk0 = nb0;
    bk1 = nb1;
  }
  // lane-reduce row max within each 16-lane half
#pragma unroll
  for (int r = 0; r < 8; ++r) {
    float v = runMax[r];
    v = fmaxf(v, __shfl_xor(v, 1, 32));
    v = fmaxf(v, __shfl_xor(v, 2, 32));
    v = fmaxf(v, __shfl_xor(v, 4, 32));
    v = fmaxf(v, __shfl_xor(v, 8, 32));
    runMax[r] = v;
  }
  if ((lane & 15) == 0) {
#pragma unroll
    for (int r = 0; r < 8; ++r) redMax[(r + 8 * half) * 4 + w] = runMax[r];
  }
  __syncthreads();
  if (threadIdx.x < 16) {
    float m = redMax[threadIdx.x * 4];
#pragma unroll
    for (int i = 1; i < 4; ++i) m = fmaxf(m, redMax[threadIdx.x * 4 + i]);
    rowMax[threadIdx.x] = m;
  }
  __syncthreads();

  // ---- exp + row sums (128 threads: row = tid>>3, interleaved chunks) ----
  {
    const int m = threadIdx.x >> 3;
    const int cch = threadIdx.x & 7;
    const float rmax = rowMax[m];
    float s = 0.f;
    for (int i = 0; i < 256; ++i) {
      const int col = cch + (i << 3);
      const float e = __expf(probs32[m * PS32 + col] - rmax);
      probs32[m * PS32 + col] = e;
      s += e;
    }
    redSum[m * 8 + cch] = s;
  }
  __syncthreads();
  if (threadIdx.x < 16) {
    float s = 0.f;
#pragma unroll
    for (int i = 0; i < 8; ++i) s += redSum[threadIdx.x * 8 + i];
    rowInv[threadIdx.x] = 1.0f / s;
  }
  __syncthreads();

  // ---- normalize: fp32 probs -> d_out, f16 probs -> LDS ----
  {
    const int m = threadIdx.x >> 3;
    const int cch = threadIdx.x & 7;
    const float inv = rowInv[m];
    float* orow = attn_out + ((size_t)bh * S_ + (s0 + m)) * S_;
    for (int i = 0; i < 256; ++i) {
      const int col = cch + (i << 3);
      const float p = probs32[m * PS32 + col] * inv;
      orow[col] = p;
      probs16[m * PS16 + col] = (_Float16)p;
    }
  }
  __syncthreads();

  // ---- out_tile = P(16x2048) @ V(2048x64); wave w owns 16 cols of d ----
  {
    const int n0 = w * 16;
    v8f c = {};
    v16h a_n = *(const v16h*)(probs16 + (size_t)(lane & 15) * PS16
                                      + ((lane >> 4) << 4));
    v16h b_n = frag_f16(vbase, S_, n0, 0, lane);
    for (int kt = 0; kt < 64; ++kt) {
      const v16h a = a_n;
      const v16h bv = b_n;
      if (kt + 1 < 64) {
        const int k0 = (kt + 1) * 32;
        a_n = *(const v16h*)(probs16 + (size_t)(lane & 15) * PS16
                                     + k0 + ((lane >> 4) << 4));
        b_n = frag_f16(vbase, S_, n0, k0, lane);     // V^T rows = dv
      }
      c = WMMA_F16(a, bv, c);
    }
    const int col = n0 + (lane & 15);
#pragma unroll
    for (int r = 0; r < 8; ++r) {
      const int m = r + 8 * half;
      ctx[((size_t)b * S_ + (s0 + m)) * D_ + (bh & 15) * DK_ + col] = (_Float16)c[r];
    }
  }
}

// ---------------------------------------------------------------------------
// Kernel C: y = LN(ctx @ w_fc + b_fc + residual)
// block = 256 threads (8 waves) -> 16 rows x 1024 cols; grid = 512
// ---------------------------------------------------------------------------
__global__ void proj_ln_kernel(const _Float16* __restrict__ ctx,
                               const _Float16* __restrict__ wT,
                               const float* __restrict__ bias,
                               const float* __restrict__ resid,
                               const float* __restrict__ gamma,
                               const float* __restrict__ beta,
                               float* __restrict__ y) {
  __shared__ float redS[16][8];
  __shared__ float redQ[16][8];
  __shared__ float rowMu[16];
  __shared__ float rowIs[16];

  const int w    = threadIdx.x >> 5;
  const int lane = threadIdx.x & 31;
  const int half = lane >> 4;
  const int m0   = blockIdx.x * 16;

  v8f acc[8] = {};
  for (int kt = 0; kt < 32; ++kt) {
    const int k0 = kt * 32;
    const v16h a = frag_f16(ctx, D_, m0, k0, lane);
    v16h bf[8];                                     // batch loads -> one clause
#pragma unroll
    for (int j = 0; j < 8; ++j)
      bf[j] = frag_f16(wT, D_, w * 128 + j * 16, k0, lane);
#pragma unroll
    for (int j = 0; j < 8; ++j)
      acc[j] = WMMA_F16(a, bf[j], acc[j]);
  }

  // bias + residual, per-lane partial sums
  float ps[8], pq[8];
#pragma unroll
  for (int r = 0; r < 8; ++r) { ps[r] = 0.f; pq[r] = 0.f; }
#pragma unroll
  for (int j = 0; j < 8; ++j) {
    const int n = w * 128 + j * 16 + (lane & 15);
    const float bv = bias[n];
#pragma unroll
    for (int r = 0; r < 8; ++r) {
      const int m = m0 + r + 8 * half;
      float x = acc[j][r] + bv + resid[(size_t)m * D_ + n];
      acc[j][r] = x;
      ps[r] += x;
      pq[r] += x * x;
    }
  }
#pragma unroll
  for (int r = 0; r < 8; ++r) {
    float s = ps[r], q = pq[r];
    s += __shfl_xor(s, 1, 32);  q += __shfl_xor(q, 1, 32);
    s += __shfl_xor(s, 2, 32);  q += __shfl_xor(q, 2, 32);
    s += __shfl_xor(s, 4, 32);  q += __shfl_xor(q, 4, 32);
    s += __shfl_xor(s, 8, 32);  q += __shfl_xor(q, 8, 32);
    ps[r] = s; pq[r] = q;
  }
  if ((lane & 15) == 0) {
#pragma unroll
    for (int r = 0; r < 8; ++r) {
      redS[r + 8 * half][w] = ps[r];
      redQ[r + 8 * half][w] = pq[r];
    }
  }
  __syncthreads();
  if (threadIdx.x < 16) {
    float s = 0.f, q = 0.f;
#pragma unroll
    for (int i = 0; i < 8; ++i) { s += redS[threadIdx.x][i]; q += redQ[threadIdx.x][i]; }
    const float mu  = s * (1.0f / D_);
    const float var = q * (1.0f / D_) - mu * mu;
    rowMu[threadIdx.x] = mu;
    rowIs[threadIdx.x] = rsqrtf(var + 1e-5f);
  }
  __syncthreads();
#pragma unroll
  for (int j = 0; j < 8; ++j) {
    const int n = w * 128 + j * 16 + (lane & 15);
    const float g = gamma[n], be = beta[n];
#pragma unroll
    for (int r = 0; r < 8; ++r) {
      const int ml = r + 8 * half;
      y[(size_t)(m0 + ml) * D_ + n] = (acc[j][r] - rowMu[ml]) * rowIs[ml] * g + be;
    }
  }
}

// ---------------------------------------------------------------------------
// launch
// ---------------------------------------------------------------------------
extern "C" void kernel_launch(void* const* d_in, const int* in_sizes, int n_in,
                              void* d_out, int out_size, void* d_ws, size_t ws_size,
                              hipStream_t stream) {
  const float* q    = (const float*)d_in[0];
  const float* k    = (const float*)d_in[1];
  const float* v    = (const float*)d_in[2];
  const int*   mask = (const int*)  d_in[3];
  const float* w_q  = (const float*)d_in[4];
  const float* b_q  = (const float*)d_in[5];
  const float* w_k  = (const float*)d_in[6];
  const float* b_k  = (const float*)d_in[7];
  const float* w_v  = (const float*)d_in[8];
  const float* b_v  = (const float*)d_in[9];
  const float* w_fc = (const float*)d_in[10];
  const float* b_fc = (const float*)d_in[11];
  const float* ln_g = (const float*)d_in[12];
  const float* ln_b = (const float*)d_in[13];

  float* y_out    = (float*)d_out;                              // [B,S,D]
  float* attn_out = y_out + (size_t)B_ * S_ * D_;               // [B,H,S,S]

  // workspace partition (f16)
  char* ws = (char*)d_ws;
  const size_t WSZ = (size_t)D_ * D_ * 2;                       // 2 MB
  _Float16* wqT  = (_Float16*)(ws + 0 * WSZ);
  _Float16* wkT  = (_Float16*)(ws + 1 * WSZ);
  _Float16* wvT  = (_Float16*)(ws + 2 * WSZ);
  _Float16* wfcT = (_Float16*)(ws + 3 * WSZ);
  char* p = ws + 4 * WSZ;
  const size_t HSZ = (size_t)B_ * H_ * S_ * DK_ * 2;            // 16 MB each
  _Float16* qh  = (_Float16*)(p);            p += HSZ;
  _Float16* kh  = (_Float16*)(p);            p += HSZ;
  _Float16* vt  = (_Float16*)(p);            p += HSZ;          // [b][h][d][S]
  _Float16* ctx = (_Float16*)(p);                               // [b][s][H*d]

  // 1) weight prep (convert + transpose to f16 [N][K])
  prep_w_kernel<<<4096, 256, 0, stream>>>(w_q,  wqT);
  prep_w_kernel<<<4096, 256, 0, stream>>>(w_k,  wkT);
  prep_w_kernel<<<4096, 256, 0, stream>>>(w_v,  wvT);
  prep_w_kernel<<<4096, 256, 0, stream>>>(w_fc, wfcT);

  // 2) QKV projections
  dim3 gA(128, 16);
  gemm_qkv_kernel<<<gA, 128, 0, stream>>>(q, wqT, b_q, qh, DK_, 1);
  gemm_qkv_kernel<<<gA, 128, 0, stream>>>(k, wkT, b_k, kh, DK_, 1);
  gemm_qkv_kernel<<<gA, 128, 0, stream>>>(v, wvT, b_v, vt, 1, S_);

  // 3) attention (scores + softmax + attn output + attn@V)
  dim3 gB(S_ / 16, B_ * H_);
  attn_kernel<<<gB, 128, SMEM_B, stream>>>(qh, kh, vt, mask, attn_out, ctx);

  // 4) output projection + residual + LayerNorm
  proj_ln_kernel<<<512, 256, 0, stream>>>(ctx, wfcT, b_fc, q, ln_g, ln_b, y_out);
}
